// SupSCN_71811853189562
// MI455X (gfx1250) — compile-verified, matrix operand order checked by reading, not run on hardware
//
#include <hip/hip_runtime.h>

// ---------------- constants (match reference) ----------------
#define NSIMP 400000
#define NEDGE 2400000
#define F_IN  32
#define HDIM  64
#define KAPPA 3
#define NGRAPH 512
#define NOUT  8
#define EPS_BN 1e-5f
#define KMAX  (7 * HDIM)        // 448 max concatenated K
#define KPAD  8                 // LDS row padding (halves)

typedef __attribute__((ext_vector_type(16))) _Float16 v16h;
typedef __attribute__((ext_vector_type(8)))  _Float16 v8h;
typedef __attribute__((ext_vector_type(8)))  float    v8f;

// ---------------- small helpers ----------------
__device__ __forceinline__ float elu1(float x) {
    return x > 0.0f ? x : (expf(x) - 1.0f);
}
// order-preserving float<->uint encoding for atomicMax on floats
__device__ __forceinline__ unsigned fenc(float f) {
    unsigned u = __float_as_uint(f);
    return (u & 0x80000000u) ? ~u : (u | 0x80000000u);
}
__device__ __forceinline__ float fdec(unsigned u) {
    return (u & 0x80000000u) ? __uint_as_float(u & 0x7FFFFFFFu)
                             : __uint_as_float(~u);
}

// ---------------- elementwise kernels ----------------
__global__ void k_zero_f32(float* p, long long n) {
    long long i = blockIdx.x * (long long)blockDim.x + threadIdx.x;
    long long s = (long long)gridDim.x * blockDim.x;
    for (; i < n; i += s) p[i] = 0.0f;
}

__global__ void k_f32_to_f16(_Float16* dst, const float* src, long long n) {
    long long i = blockIdx.x * (long long)blockDim.x + threadIdx.x;
    long long s = (long long)gridDim.x * blockDim.x;
    for (; i < n; i += s) dst[i] = (_Float16)src[i];
}

// convert a dense [nrows x F] f32 buffer into a column-slice of xcat
// (row stride = Ktot halves):  dst[r*stride + f] = src[r*F + f]
__global__ void k_f32_to_f16_slice(_Float16* dst, const float* src,
                                   long long nrows, int lf, int stride) {
    long long n = nrows << lf;
    long long msk = (1ll << lf) - 1;
    long long i = blockIdx.x * (long long)blockDim.x + threadIdx.x;
    long long s = (long long)gridDim.x * blockDim.x;
    for (; i < n; i += s) {
        long long r = i >> lf;
        int f = (int)(i & msk);
        dst[r * (long long)stride + f] = (_Float16)src[i];
    }
}

// ---------------- SpMM: y[i] += val_e * x[j],  i=ei[0][e], j=ei[1][e] ----------------
// one thread per (edge, 4-float chunk);  F = 32 or 64, lc = log2(F/4)
__global__ void k_spmm_atomic(float* __restrict__ y, const int* __restrict__ ei,
                              const float* __restrict__ ev,
                              const float* __restrict__ x, int F, int lc) {
    long long chunks_mask = (1ll << lc) - 1;
    long long total = (long long)NEDGE << lc;
    long long idx = blockIdx.x * (long long)blockDim.x + threadIdx.x;
    if (idx >= total) return;
    int e = (int)(idx >> lc);
    int c = (int)(idx & chunks_mask) * 4;
    int i = ei[e];
    int j = ei[NEDGE + e];
    float v = ev[e];
    const float4 xv = *(const float4*)&x[(long long)j * F + c];
    float* yp = &y[(long long)i * F + c];
    atomicAdd(yp + 0, v * xv.x);
    atomicAdd(yp + 1, v * xv.y);
    atomicAdd(yp + 2, v * xv.z);
    atomicAdd(yp + 3, v * xv.w);
}

// ---------------- fused layer GEMM (WMMA) ----------------
// z[N x 64] = bias_total + A[N x Ktot] @ W[Ktot x 64]
// block = 256 threads (8 wave32s); block tile = 128 rows x 64 cols;
// wave w computes rows [w*16, w*16+16) x all 4 col tiles.
// W is transposed into LDS with padded stride to kill bank conflicts, so the
// per-lane B fragment (fixed n, contiguous k per ISA 16-bit layout) is two
// 16-byte ds loads.
__global__ void __launch_bounds__(256)
k_gemm_layer_f16(float* __restrict__ z, const _Float16* __restrict__ A,
                 const _Float16* __restrict__ W,
                 const float* __restrict__ bh, const float* __restrict__ bl,
                 const float* __restrict__ bu, int Ktot) {
    __shared__ __align__(16) _Float16 lwt[HDIM * (KMAX + KPAD)];  // 58.4 KB max
    const int KP = Ktot + KPAD;
    const int tid = threadIdx.x;
    // transpose W[Ktot x 64] -> lwt[n*KP + k] (coalesced global reads)
    for (int i = tid; i < Ktot * HDIM; i += 256) {
        int k = i >> 6, n = i & 63;
        lwt[n * KP + k] = W[i];
    }
    __syncthreads();

    const int wave = tid >> 5, lane = tid & 31;
    const int m = lane & 15, hi = lane >> 4;
    const int row0 = blockIdx.x * 128 + wave * 16;

    // C init = total bias (bh + sum_k bl + bu) broadcast per column
    v8f acc[4];
#pragma unroll
    for (int ct = 0; ct < 4; ++ct) {
        int col = ct * 16 + m;
        float b = bh[col];
#pragma unroll
        for (int k = 0; k < KAPPA; ++k) b += bl[k * HDIM + col] + bu[k * HDIM + col];
#pragma unroll
        for (int v = 0; v < 8; ++v) acc[ct][v] = b;
    }

    const _Float16* arow = &A[(long long)(row0 + m) * Ktot];
    for (int kb = 0; kb < Ktot; kb += 32) {
        // A fragment (16x32 f16): lane holds row m, K = kb + hi*8 + {0..7} and +16
        union { v16h v; v8h h2[2]; } a;
        const _Float16* ap = arow + kb + hi * 8;
        __builtin_prefetch((const void*)(ap + 32), 0, 3);  // next K chunk
        a.h2[0] = *(const v8h*)(ap);
        a.h2[1] = *(const v8h*)(ap + 16);
#pragma unroll
        for (int ct = 0; ct < 4; ++ct) {
            // B fragment: lane holds column n = ct*16+m, same K pattern
            union { v16h v; v8h h2[2]; } b;
            const _Float16* bp = &lwt[(ct * 16 + m) * KP + kb + hi * 8];
            b.h2[0] = *(const v8h*)(bp);
            b.h2[1] = *(const v8h*)(bp + 16);
            acc[ct] = __builtin_amdgcn_wmma_f32_16x16x32_f16(
                /*neg_a=*/false, a.v, /*neg_b=*/false, b.v,
                /*c_mod=*/(short)0, acc[ct], /*reuse_a=*/false, /*reuse_b=*/false);
        }
    }

    // C/D layout: VGPR v -> (lanes 0-15: M=v, N=lane) (lanes 16-31: M=v+8, N=lane-16)
    const int crow = row0 + hi * 8;
#pragma unroll
    for (int ct = 0; ct < 4; ++ct)
#pragma unroll
        for (int v = 0; v < 8; ++v)
            z[(long long)(crow + v) * HDIM + ct * 16 + m] = acc[ct][v];
}

// ---------------- BatchNorm (training stats over N rows) ----------------
__global__ void k_bn_stats(const float* __restrict__ z, float* __restrict__ stats,
                           long long nrows) {
    int col = threadIdx.x & (HDIM - 1);
    int sub = threadIdx.x >> 6;  // 0..3 for 256 threads
    long long r = (long long)blockIdx.x * 4 + sub;
    long long stride = (long long)gridDim.x * 4;
    float s = 0.0f, q = 0.0f;
    for (; r < nrows; r += stride) {
        float v = z[r * HDIM + col];
        s += v;
        q += v * v;
    }
    atomicAdd(&stats[col], s);
    atomicAdd(&stats[HDIM + col], q);
}

__global__ void k_bn_apply_elu(float* z, const float* __restrict__ stats,
                               const float* __restrict__ g,
                               const float* __restrict__ b, long long nrows) {
    long long n = nrows * HDIM;
    long long i = blockIdx.x * (long long)blockDim.x + threadIdx.x;
    long long s = (long long)gridDim.x * blockDim.x;
    float inv_n = 1.0f / (float)nrows;
    for (; i < n; i += s) {
        int h = (int)(i & (HDIM - 1));
        float mean = stats[h] * inv_n;
        float var = stats[HDIM + h] * inv_n - mean * mean;
        float r = rsqrtf(var + EPS_BN);
        float v = (z[i] - mean) * r * g[h] + b[h];
        z[i] = elu1(v);
    }
}

// ---------------- pooling ----------------
__global__ void k_pool_init(unsigned* pmax, float* psum, float* pcnt) {
    int i = blockIdx.x * blockDim.x + threadIdx.x;
    if (i < NGRAPH * HDIM) {
        pmax[i] = 0x007FFFFFu;  // fenc(-inf)
        psum[i] = 0.0f;
    }
    if (i < NGRAPH) pcnt[i] = 0.0f;
}

__global__ void k_pool_scatter(const float* __restrict__ h,
                               const int* __restrict__ bi, unsigned* pmax,
                               float* psum, float* pcnt, long long nrows) {
    long long n = nrows * HDIM;
    long long i = blockIdx.x * (long long)blockDim.x + threadIdx.x;
    long long s = (long long)gridDim.x * blockDim.x;
    for (; i < n; i += s) {
        long long r = i >> 6;
        int c = (int)(i & (HDIM - 1));
        int g = bi[r];
        float v = h[i];
        atomicMax(&pmax[g * HDIM + c], fenc(v));
        atomicAdd(&psum[g * HDIM + c], v);
        if (c == 0) atomicAdd(&pcnt[g], 1.0f);
    }
}

// ---------------- head: elu(concat[max,mean]) -> relu MLP -> out ----------------
__global__ void __launch_bounds__(64)
k_head_mlp(const unsigned* __restrict__ pmax, const float* __restrict__ psum,
           const float* __restrict__ pcnt, const float* __restrict__ W1,
           const float* __restrict__ b1, const float* __restrict__ W2,
           const float* __restrict__ b2, float* __restrict__ out) {
    __shared__ float gc[2 * HDIM];
    __shared__ float hid[HDIM];
    int g = blockIdx.x, t = threadIdx.x;
    float cnt = fmaxf(pcnt[g], 1.0f);
    gc[t]        = elu1(fdec(pmax[g * HDIM + t]));
    gc[HDIM + t] = elu1(psum[g * HDIM + t] / cnt);
    __syncthreads();
    float acc = b1[t];
#pragma unroll 8
    for (int j = 0; j < 2 * HDIM; ++j) acc += gc[j] * W1[j * HDIM + t];
    hid[t] = fmaxf(acc, 0.0f);
    __syncthreads();
    if (t < NOUT) {
        float o = b2[t];
#pragma unroll 8
        for (int k = 0; k < HDIM; ++k) o += hid[k] * W2[k * NOUT + t];
        out[g * NOUT + t] = o;
    }
}

// ---------------- host orchestration ----------------
extern "C" void kernel_launch(void* const* d_in, const int* in_sizes, int n_in,
                              void* d_out, int out_size, void* d_ws, size_t ws_size,
                              hipStream_t stream) {
    (void)in_sizes; (void)n_in; (void)out_size; (void)ws_size;
    const float* x   = (const float*)d_in[0];
    const int*   li  = (const int*)d_in[1];
    const float* lv  = (const float*)d_in[2];
    const int*   ui  = (const int*)d_in[3];
    const float* uv  = (const float*)d_in[4];
    const int*   bi  = (const int*)d_in[5];
    const float* l0_Wl = (const float*)d_in[6];
    const float* l0_bl = (const float*)d_in[7];
    const float* l0_Wu = (const float*)d_in[8];
    const float* l0_bu = (const float*)d_in[9];
    const float* l0_Wh = (const float*)d_in[10];
    const float* l0_bh = (const float*)d_in[11];
    const float* l1_Wl = (const float*)d_in[12];
    const float* l1_bl = (const float*)d_in[13];
    const float* l1_Wu = (const float*)d_in[14];
    const float* l1_bu = (const float*)d_in[15];
    const float* l1_Wh = (const float*)d_in[16];
    const float* l1_bh = (const float*)d_in[17];
    const float* bn0_g = (const float*)d_in[18];
    const float* bn0_b = (const float*)d_in[19];
    const float* bn1_g = (const float*)d_in[20];
    const float* bn1_b = (const float*)d_in[21];
    const float* mlp_W1 = (const float*)d_in[22];
    const float* mlp_b1 = (const float*)d_in[23];
    const float* mlp_W2 = (const float*)d_in[24];
    const float* mlp_b2 = (const float*)d_in[25];
    float* out = (float*)d_out;

    const long long N = NSIMP;

    // ---- workspace carve ----
    char* ws = (char*)d_ws;
    size_t off = 0;
    auto carve = [&](size_t bytes) {
        void* p = ws + off;
        off += (bytes + 255) & ~(size_t)255;
        return p;
    };
    float*    bufA  = (float*)carve((size_t)N * HDIM * 4);
    float*    bufB  = (float*)carve((size_t)N * HDIM * 4);
    float*    z0    = (float*)carve((size_t)N * HDIM * 4);
    float*    z1    = (float*)carve((size_t)N * HDIM * 4);
    _Float16* xcat  = (_Float16*)carve((size_t)N * KMAX * 2);   // concat features
    _Float16* wcat  = (_Float16*)carve((size_t)KMAX * HDIM * 2);
    float*    stats = (float*)carve(2 * HDIM * 4);
    unsigned* pmax  = (unsigned*)carve((size_t)NGRAPH * HDIM * 4);
    float*    psum  = (float*)carve((size_t)NGRAPH * HDIM * 4);
    float*    pcnt  = (float*)carve((size_t)NGRAPH * 4);

    const int EW = 256;   // elementwise block
    const int EG = 4096;  // elementwise grid (grid-stride)
    const int gemm_blocks = NSIMP / 128;  // 3125

    auto run_layer = [&](const float* xin, int F, int lf,
                         const float* Whf, const float* Wlf, const float* Wuf,
                         const float* bh, const float* bl, const float* bu,
                         float* z) {
        const int Ktot = 7 * F;
        const int lc = lf - 2;  // log2(F/4)
        long long spmm_threads = (long long)NEDGE << lc;
        int spmm_blocks = (int)((spmm_threads + EW - 1) / EW);

        // weights -> wcat (rows: [Wh (F) | Wl (3F) | Wu (3F)] x 64)
        k_f32_to_f16<<<32, EW, 0, stream>>>(wcat, Whf, (long long)F * HDIM);
        k_f32_to_f16<<<96, EW, 0, stream>>>(wcat + (size_t)F * HDIM, Wlf,
                                            (long long)3 * F * HDIM);
        k_f32_to_f16<<<96, EW, 0, stream>>>(wcat + (size_t)4 * F * HDIM, Wuf,
                                            (long long)3 * F * HDIM);
        // slice 0 = x
        k_f32_to_f16_slice<<<EG, EW, 0, stream>>>(xcat, xin, N, lf, Ktot);
        // Laplacian power chains -> slices 1..3 (lower), 4..6 (upper)
        for (int side = 0; side < 2; ++side) {
            const int*   ei = side ? ui : li;
            const float* ev = side ? uv : lv;
            const float* cur = xin;
            for (int k = 0; k < KAPPA; ++k) {
                float* nxt = (k & 1) ? bufB : bufA;
                k_zero_f32<<<EG, EW, 0, stream>>>(nxt, N * (long long)F);
                k_spmm_atomic<<<spmm_blocks, EW, 0, stream>>>(nxt, ei, ev, cur, F, lc);
                int slice = 1 + side * KAPPA + k;
                k_f32_to_f16_slice<<<EG, EW, 0, stream>>>(
                    xcat + (size_t)slice * F, nxt, N, lf, Ktot);
                cur = nxt;
            }
        }
        // one fused WMMA GEMM: z = bias_total + xcat @ wcat
        k_gemm_layer_f16<<<gemm_blocks, 256, 0, stream>>>(z, xcat, wcat,
                                                          bh, bl, bu, Ktot);
    };

    // ---- layer 0 (F_IN=32 -> 64), BN0 + ELU in place on z0 ----
    run_layer(x, F_IN, 5, l0_Wh, l0_Wl, l0_Wu, l0_bh, l0_bl, l0_bu, z0);
    k_zero_f32<<<1, 128, 0, stream>>>(stats, 2 * HDIM);
    k_bn_stats<<<1024, 256, 0, stream>>>(z0, stats, N);
    k_bn_apply_elu<<<EG, EW, 0, stream>>>(z0, stats, bn0_g, bn0_b, N);

    // ---- layer 1 (64 -> 64), BN1 + ELU in place on z1 ----
    run_layer(z0, HDIM, 6, l1_Wh, l1_Wl, l1_Wu, l1_bh, l1_bl, l1_bu, z1);
    k_zero_f32<<<1, 128, 0, stream>>>(stats, 2 * HDIM);
    k_bn_stats<<<1024, 256, 0, stream>>>(z1, stats, N);
    k_bn_apply_elu<<<EG, EW, 0, stream>>>(z1, stats, bn1_g, bn1_b, N);

    // ---- pooling + head MLP ----
    k_pool_init<<<(NGRAPH * HDIM + EW - 1) / EW, EW, 0, stream>>>(pmax, psum, pcnt);
    k_pool_scatter<<<EG, EW, 0, stream>>>(z1, bi, pmax, psum, pcnt, N);
    k_head_mlp<<<NGRAPH, 64, 0, stream>>>(pmax, psum, pcnt, mlp_W1, mlp_b1,
                                          mlp_W2, mlp_b2, out);
}
